// patchViT_74345883894462
// MI455X (gfx1250) — compile-verified, matrix operand order checked by reading, not run on hardware
//
#include <hip/hip_runtime.h>

// ---------------------------------------------------------------------------
// patchViT for MI455X (gfx1250, wave32). out = x except the last 64x64 patch,
// which gets 4 layers of f-test-gated channel self-attention.
// All GEMMs use V_WMMA_F32_16X16X32_F16; each WMMA operand is two (f16) or four (f32) 16-byte
// loads per lane (K-contiguous layouts). No float atomics: all reductions are
// fixed-order for bit-stable replays.
// ---------------------------------------------------------------------------

#define BATCH  4
#define CIN    256
#define HH     256
#define WWID   256
#define PTCH   64
#define ICH    64
#define PP     4096
#define HWTOT  65536

typedef __attribute__((ext_vector_type(16))) _Float16 v16h;
typedef __attribute__((ext_vector_type(8)))  _Float16 v8h;
typedef __attribute__((ext_vector_type(8)))  float    v8f;
typedef __attribute__((ext_vector_type(4)))  float    f4;

#define WMMA16(a, b, c) \
  __builtin_amdgcn_wmma_f32_16x16x32_f16(false, (a), false, (b), (short)0, (c), false, false)

// workspace layout (float units)
static constexpr size_t OFF_XH    = 0;                     // 1-batch NHWC f16 image
static constexpr size_t OFF_PT0   = 8388608;               // patch f16 (pix,ch) ping
static constexpr size_t OFF_PT1   = OFF_PT0 + 2097152;     // pong
static constexpr size_t OFF_Q     = OFF_PT1 + 2097152;     // Q f16 (b,64,4096) row-major
static constexpr size_t OFF_KT    = OFF_Q   + 524288;      // transpose of flat-view K
static constexpr size_t OFF_VT    = OFF_KT  + 524288;      // V^T f16 (b,4096,64)
static constexpr size_t OFF_SCP   = OFF_VT  + 524288;      // 8 split-K partial scores f32
static constexpr size_t OFF_SC    = OFF_SCP + 131072;      // scores f32 (b,64,64)
static constexpr size_t OFF_WTST  = OFF_SC  + 16384;       // wts^T f16
static constexpr size_t OFF_MH    = OFF_WTST + 8192;       // M f16 (b,256,64) row-major
static constexpr size_t OFF_PART2 = OFF_MH  + 32768;       // full-image partials 4096+4096
static constexpr size_t OFF_PART1 = OFF_PART2 + 8192;      // patch partials 1024+1024
static constexpr size_t OFF_ST2   = OFF_PART1 + 2048;      // m2[64], v2[64]
static constexpr size_t OFF_FLAG  = OFF_ST2 + 128;         // stopped flag

// ----------------------- tile loaders: 2 (f16) / 4 (f32) x b128 per lane --------------------
__device__ __forceinline__ v16h ld_tile_f16(const _Float16* __restrict__ T, int ld, int lane) {
  int m = lane & 15, kh = lane >> 4;
  const _Float16* p = T + (size_t)m * ld + kh * 8;
  v8h lo = *(const v8h*)p;
  v8h hi = *(const v8h*)(p + 16);
  v16h a;
#pragma unroll
  for (int i = 0; i < 8; ++i) { a[i] = lo[i]; a[8 + i] = hi[i]; }
  return a;
}
__device__ __forceinline__ v16h ld_tile_f32(const float* __restrict__ T, int ld, int lane) {
  int m = lane & 15, kh = lane >> 4;
  const float* p = T + (size_t)m * ld + kh * 8;
  f4 a0 = *(const f4*)p;
  f4 a1 = *(const f4*)(p + 4);
  f4 b0 = *(const f4*)(p + 16);
  f4 b1 = *(const f4*)(p + 20);
  v16h a;
#pragma unroll
  for (int i = 0; i < 4; ++i) {
    a[i]      = (_Float16)a0[i];
    a[4 + i]  = (_Float16)a1[i];
    a[8 + i]  = (_Float16)b0[i];
    a[12 + i] = (_Float16)b1[i];
  }
  return a;
}

// ------------------------------- kernels -----------------------------------
__global__ void k_zero(float* __restrict__ ws) {
  if (threadIdx.x == 0) *(int*)(ws + OFF_FLAG) = 0;
}

// out = x for one batch + f16 NHWC staging image for the f-test GEMM
__global__ void k_copy_xpose(const float* __restrict__ x, float* __restrict__ out,
                             _Float16* __restrict__ xh, int b) {
  __shared__ _Float16 lds[64 * CIN];        // 32KB
  int t = threadIdx.x;                      // channel
  int row = blockIdx.y, ph = blockIdx.x;    // 4 x 64-pixel spans per row
  size_t base = (((size_t)(b * CIN + t) * HH) + row) * WWID + (size_t)ph * 64;
  const f4* src = (const f4*)(x + base);
  f4* dst = (f4*)(out + base);
#pragma unroll
  for (int i = 0; i < 16; ++i) {
    f4 v = src[i];
    dst[i] = v;
#pragma unroll
    for (int e = 0; e < 4; ++e) lds[(i * 4 + e) * CIN + t] = (_Float16)v[e];
  }
  __syncthreads();
  v8h* od = (v8h*)(xh + ((size_t)row * WWID + (size_t)ph * 64) * CIN);
  const v8h* ls = (const v8h*)lds;
#pragma unroll
  for (int i = 0; i < 8; ++i) od[t * 8 + i] = ls[t * 8 + i];
}

// extract last patch -> f16 (pix, ch)
__global__ void k_extract(const float* __restrict__ x, _Float16* __restrict__ Pt) {
  __shared__ _Float16 lds[PTCH * CIN];      // 32KB
  int t = threadIdx.x, r = blockIdx.x, b = blockIdx.y;
  const f4* src = (const f4*)(x + (((size_t)(b * CIN + t) * HH + 192 + r) * WWID + 192));
#pragma unroll
  for (int i = 0; i < 16; ++i) {
    f4 v = src[i];
#pragma unroll
    for (int e = 0; e < 4; ++e) lds[(i * 4 + e) * CIN + t] = (_Float16)v[e];
  }
  __syncthreads();
  v8h* dst = (v8h*)(Pt + ((size_t)b * PP + (size_t)r * PTCH) * CIN);
  const v8h* ls = (const v8h*)lds;
#pragma unroll
  for (int i = 0; i < 8; ++i) dst[t * 8 + i] = ls[t * 8 + i];
}

// full-image f-test stats, one batch per launch, per-chunk partials
__global__ void __launch_bounds__(128, 1)
k_fullstats(const _Float16* __restrict__ xh, const float* __restrict__ Wf,
            const float* __restrict__ bf, float* __restrict__ ws, int b) {
  int w = threadIdx.x >> 5, lane = threadIdx.x & 31;
  int chunk = blockIdx.x * 4 + w;           // 0..63, each 64 pixel-tiles
  int mh = lane >> 4;
  v16h A[8];
#pragma unroll
  for (int kk = 0; kk < 8; ++kk) A[kk] = ld_tile_f32(Wf + kk * 32, CIN, lane);
  float ls[8], lq[8];
#pragma unroll
  for (int r = 0; r < 8; ++r) { ls[r] = 0.f; lq[r] = 0.f; }
  const _Float16* xp = xh + (size_t)chunk * (64 * 16 * CIN);
#pragma unroll 1
  for (int t = 0; t < 64; ++t) {
    v8f c = {};
#pragma unroll
    for (int kk = 0; kk < 8; ++kk) {
      v16h bm = ld_tile_f16(xp + kk * 32, CIN, lane);
      c = WMMA16(A[kk], bm, c);
    }
    xp += 16 * CIN;
#pragma unroll
    for (int r = 0; r < 8; ++r) {
      float val = c[r] + bf[mh * 8 + r];
      ls[r] += val; lq[r] += val * val;
    }
  }
#pragma unroll
  for (int r = 0; r < 8; ++r)
    for (int m = 1; m <= 8; m <<= 1) {
      ls[r] += __shfl_xor(ls[r], m, 32);
      lq[r] += __shfl_xor(lq[r], m, 32);
    }
  if ((lane & 15) == 0) {
    float* ps = ws + OFF_PART2 + (size_t)(b * 64 + chunk) * 16;
#pragma unroll
    for (int r = 0; r < 8; ++r) {
      int o = mh * 8 + r;
      ps[o] = ls[r];
      ps[4096 + o] = lq[r];
    }
  }
}

// fixed-order reduction of full-image partials -> m2, v2
__global__ void k_finalize(float* __restrict__ ws) {
  int i = threadIdx.x;                      // 64 = b*16 + o
  int b = i >> 4, o = i & 15;
  float s = 0.f, q = 0.f;
  for (int ch = 0; ch < 64; ++ch) {
    size_t idx = OFF_PART2 + (size_t)(b * 64 + ch) * 16 + o;
    s += ws[idx];
    q += ws[idx + 4096];
  }
  float m = s / (float)HWTOT;
  ws[OFF_ST2 + i] = m;
  ws[OFF_ST2 + 64 + i] = q / (float)HWTOT - m * m;
}

// per-layer patch f-test stats, per-chunk partials
__global__ void __launch_bounds__(128, 1)
k_patchstats(const _Float16* __restrict__ Pt, const float* __restrict__ Wf,
             const float* __restrict__ bf, float* __restrict__ ws) {
  int w = threadIdx.x >> 5, lane = threadIdx.x & 31;
  int chunk = blockIdx.x * 4 + w;           // 0..15, each 16 pixel-tiles
  int b = blockIdx.y;
  int mh = lane >> 4;
  v16h A[8];
#pragma unroll
  for (int kk = 0; kk < 8; ++kk) A[kk] = ld_tile_f32(Wf + kk * 32, CIN, lane);
  float ls[8], lq[8];
#pragma unroll
  for (int r = 0; r < 8; ++r) { ls[r] = 0.f; lq[r] = 0.f; }
  const _Float16* pp = Pt + ((size_t)b * PP + (size_t)chunk * 256) * CIN;
#pragma unroll 1
  for (int t = 0; t < 16; ++t) {
    v8f c = {};
#pragma unroll
    for (int kk = 0; kk < 8; ++kk) {
      v16h bm = ld_tile_f16(pp + kk * 32, CIN, lane);
      c = WMMA16(A[kk], bm, c);
    }
    pp += 16 * CIN;
#pragma unroll
    for (int r = 0; r < 8; ++r) {
      float val = c[r] + bf[mh * 8 + r];
      ls[r] += val; lq[r] += val * val;
    }
  }
#pragma unroll
  for (int r = 0; r < 8; ++r)
    for (int m = 1; m <= 8; m <<= 1) {
      ls[r] += __shfl_xor(ls[r], m, 32);
      lq[r] += __shfl_xor(lq[r], m, 32);
    }
  if ((lane & 15) == 0) {
    float* ps = ws + OFF_PART1 + (size_t)(b * 16 + chunk) * 16;
#pragma unroll
    for (int r = 0; r < 8; ++r) {
      int o = mh * 8 + r;
      ps[o] = ls[r];
      ps[1024 + o] = lq[r];
    }
  }
}

// fixed-order reduce + ANOVA gate: pval<0.05 <=> F > F_crit(1,69630,.95)~=3.8417
__global__ void k_decide(float* __restrict__ ws) {
  __shared__ int cnt;
  int tid = threadIdx.x;                    // 64 = b*16 + o
  if (tid == 0) cnt = 0;
  __syncthreads();
  int b = tid >> 4, o = tid & 15;
  float s1 = 0.f, q1 = 0.f;
  for (int ch = 0; ch < 16; ++ch) {
    size_t idx = OFF_PART1 + (size_t)(b * 16 + ch) * 16 + o;
    s1 += ws[idx];
    q1 += ws[idx + 1024];
  }
  const float n1 = (float)PP, n2 = (float)HWTOT;
  const float dfw = n1 + n2 - 2.0f;
  float m1 = s1 / n1;
  float v1 = q1 / n1 - m1 * m1;
  float m2 = ws[OFF_ST2 + tid];
  float v2 = ws[OFF_ST2 + 64 + tid];
  float gm = (n1 * m1 + n2 * m2) / (n1 + n2);
  float ssb = n1 * (m1 - gm) * (m1 - gm) + n2 * (m2 - gm) * (m2 - gm);
  float ssw = n1 * v1 + n2 * v2;
  float F = ssb * dfw / ssw;
  if (F > 3.841707f) atomicAdd(&cnt, 1);    // integer: order-invariant
  __syncthreads();
  if (tid == 0 && 2 * cnt >= 64) *(int*)(ws + OFF_FLAG) = 1;
}

// Q/K/V convs; Q row-major, K stored as transpose of its flat (4096,64) view,
// V stored transposed -- all K-contiguous for ld_tile.
__global__ void __launch_bounds__(128, 1)
k_qkv(const _Float16* __restrict__ Pt,
      const float* __restrict__ Wq, const float* __restrict__ bq,
      const float* __restrict__ Wk, const float* __restrict__ bk,
      const float* __restrict__ Wv, const float* __restrict__ bv,
      float* __restrict__ ws) {
  int w = threadIdx.x >> 5, lane = threadIdx.x & 31;
  int b = blockIdx.z, ot = blockIdx.y;
  int col0 = (blockIdx.x * 4 + w) * 16;
  int wsel = ot >> 2, row0 = (ot & 3) * 16;
  const float* W    = (wsel == 0) ? Wq : (wsel == 1) ? Wk : Wv;
  const float* bias = (wsel == 0) ? bq : (wsel == 1) ? bk : bv;
  v8f c = {};
#pragma unroll
  for (int kk = 0; kk < 8; ++kk) {
    v16h a  = ld_tile_f32(W + row0 * CIN + kk * 32, CIN, lane);
    v16h bm = ld_tile_f16(Pt + ((size_t)b * PP + col0) * CIN + kk * 32, CIN, lane);
    c = WMMA16(a, bm, c);
  }
  int n = lane & 15, mh = lane >> 4;
  if (wsel == 2) {                           // V^T: 8 consecutive halves
    _Float16* VT = (_Float16*)(ws + OFF_VT);
    v8h pk;
#pragma unroll
    for (int r = 0; r < 8; ++r) pk[r] = (_Float16)(c[r] + bias[row0 + mh * 8 + r]);
    *(v8h*)(VT + ((size_t)b * PP + col0 + n) * ICH + row0 + mh * 8) = pk;
  } else if (wsel == 0) {                    // Q row-major
    _Float16* Q = (_Float16*)(ws + OFF_Q);
#pragma unroll
    for (int r = 0; r < 8; ++r) {
      int o = row0 + mh * 8 + r;
      Q[((size_t)b * ICH + o) * PP + col0 + n] = (_Float16)(c[r] + bias[o]);
    }
  } else {                                   // K^T of the flat (4096,64) view
    _Float16* KT = (_Float16*)(ws + OFF_KT) + (size_t)b * ICH * PP;
    int j = (col0 + n) & 63, mlow = (col0 + n) >> 6;
#pragma unroll
    for (int r = 0; r < 8; ++r) {
      int o = row0 + mh * 8 + r;
      KT[(size_t)j * PP + o * 64 + mlow] = (_Float16)(c[r] + bias[o]);
    }
  }
}

// split-K scores: partial[ks] = Q[:, ks*512:+512] @ flatK[ks*512:+512, :]
__global__ void __launch_bounds__(128, 1)
k_qk(float* __restrict__ ws) {
  int w = threadIdx.x >> 5, lane = threadIdx.x & 31;
  int b = blockIdx.y, ks = blockIdx.x, it = w;
  const _Float16* Qh = (const _Float16*)(ws + OFF_Q) +
                       ((size_t)b * ICH + it * 16) * PP + (size_t)ks * 512;
  const _Float16* KT = (const _Float16*)(ws + OFF_KT) +
                       (size_t)b * ICH * PP + (size_t)ks * 512;
  v8f c0 = {}, c1 = {}, c2 = {}, c3 = {};
#pragma unroll 1
  for (int kk = 0; kk < 16; ++kk) {
    v16h a  = ld_tile_f16(Qh + kk * 32, PP, lane);
    v16h b0 = ld_tile_f16(KT + (size_t)0 * 16 * PP + kk * 32, PP, lane);
    v16h b1 = ld_tile_f16(KT + (size_t)1 * 16 * PP + kk * 32, PP, lane);
    v16h b2 = ld_tile_f16(KT + (size_t)2 * 16 * PP + kk * 32, PP, lane);
    v16h b3 = ld_tile_f16(KT + (size_t)3 * 16 * PP + kk * 32, PP, lane);
    c0 = WMMA16(a, b0, c0);
    c1 = WMMA16(a, b1, c1);
    c2 = WMMA16(a, b2, c2);
    c3 = WMMA16(a, b3, c3);
  }
  float* sc = ws + OFF_SCP + ((size_t)ks * BATCH + b) * 4096;
  int n = lane & 15, mh = lane >> 4;
#pragma unroll
  for (int r = 0; r < 8; ++r) {
    int i = it * 16 + mh * 8 + r;
    sc[i * ICH + 0 * 16 + n] = c0[r];
    sc[i * ICH + 1 * 16 + n] = c1[r];
    sc[i * ICH + 2 * 16 + n] = c2[r];
    sc[i * ICH + 3 * 16 + n] = c3[r];
  }
}

// fixed-order split-K reduction
__global__ void k_screduce(float* __restrict__ ws) {
  int i = blockIdx.x * blockDim.x + threadIdx.x;  // 16384 = b*4096 + i*64 + j
  int b = i >> 12, r = i & 4095;
  float s = 0.f;
#pragma unroll
  for (int ks = 0; ks < 8; ++ks)
    s += ws[OFF_SCP + ((size_t)ks * BATCH + b) * 4096 + r];
  ws[OFF_SC + i] = s;
}

// softmax over rows i per column j; emit wts^T (j-major, contiguous)
__global__ void k_softmax(float* __restrict__ ws) {
  int tid = threadIdx.x;                    // 256 = (b, j)
  int b = tid >> 6, j = tid & 63;
  const float* sc = ws + OFF_SC + (size_t)b * 4096;
  float mx = -1e30f;
  for (int i = 0; i < 64; ++i) mx = fmaxf(mx, sc[i * 64 + j]);
  float sum = 0.f;
  for (int i = 0; i < 64; ++i) sum += __expf(sc[i * 64 + j] - mx);
  float inv = 1.f / sum;
  _Float16* wt = (_Float16*)(ws + OFF_WTST) + (size_t)b * 4096 + (size_t)j * 64;
  for (int i = 0; i < 64; ++i) wt[i] = (_Float16)(__expf(sc[i * 64 + j] - mx) * inv);
}

// M = Wr(256x64) @ wts(64x64); wave computes 16x64 with 4 accums
__global__ void __launch_bounds__(128, 1)
k_mmat(const float* __restrict__ Wr, float* __restrict__ ws) {
  int w = threadIdx.x >> 5, lane = threadIdx.x & 31;
  int b = blockIdx.y, rt = blockIdx.x * 4 + w;   // 0..15
  const _Float16* wt = (const _Float16*)(ws + OFF_WTST) + (size_t)b * 4096;
  v8f c[4] = {};
#pragma unroll
  for (int kk = 0; kk < 2; ++kk) {
    v16h a = ld_tile_f32(Wr + (rt * 16) * ICH + kk * 32, ICH, lane);
#pragma unroll
    for (int ct = 0; ct < 4; ++ct) {
      v16h bm = ld_tile_f16(wt + (ct * 16) * ICH + kk * 32, ICH, lane);
      c[ct] = WMMA16(a, bm, c[ct]);
    }
  }
  _Float16* M = (_Float16*)(ws + OFF_MH) + (size_t)b * CIN * ICH;
  int n = lane & 15, mh = lane >> 4;
#pragma unroll
  for (int ct = 0; ct < 4; ++ct)
#pragma unroll
    for (int r = 0; r < 8; ++r)
      M[(size_t)(rt * 16 + mh * 8 + r) * ICH + ct * 16 + n] = (_Float16)c[ct][r];
}

// Pnext = stopped ? Pcur : M(256x64)@V(64x4096) + br + Pcur   ((pix,ch) f16)
__global__ void __launch_bounds__(128, 1)
k_apply(const _Float16* __restrict__ Ptc, _Float16* __restrict__ Ptn,
        const float* __restrict__ br, float* __restrict__ ws) {
  int w = threadIdx.x >> 5, lane = threadIdx.x & 31;
  int b = blockIdx.z, rt = blockIdx.y;
  int col0 = (blockIdx.x * 4 + w) * 16;
  int n = lane & 15, mh = lane >> 4;
  size_t resoff = ((size_t)b * PP + col0 + n) * CIN + rt * 16 + mh * 8;
  const v8h* rin = (const v8h*)(Ptc + resoff);
  v8h* rout = (v8h*)(Ptn + resoff);
  int flag = *(const int*)(ws + OFF_FLAG);  // wave-uniform branch
  if (flag) { *rout = *rin; return; }
  const _Float16* M  = (const _Float16*)(ws + OFF_MH) + ((size_t)b * CIN + rt * 16) * ICH;
  const _Float16* VT = (const _Float16*)(ws + OFF_VT) + ((size_t)b * PP + col0) * ICH;
  v8f c = {};
#pragma unroll
  for (int kk = 0; kk < 2; ++kk) {
    v16h a  = ld_tile_f16(M + kk * 32, ICH, lane);
    v16h bm = ld_tile_f16(VT + kk * 32, ICH, lane);
    c = WMMA16(a, bm, c);
  }
  v8h res = *rin;
  v8h outp;
#pragma unroll
  for (int r = 0; r < 8; ++r)
    outp[r] = (_Float16)(c[r] + br[rt * 16 + mh * 8 + r] + (float)res[r]);
  *rout = outp;
}

// final patch -> f32 planar region of out
__global__ void k_writeback(const _Float16* __restrict__ Pt, float* __restrict__ out) {
  __shared__ _Float16 lds[PTCH * CIN];      // 32KB
  int t = threadIdx.x, r = blockIdx.x, b = blockIdx.y;
  const v8h* src = (const v8h*)(Pt + ((size_t)b * PP + (size_t)r * PTCH) * CIN);
  v8h* ls = (v8h*)lds;
#pragma unroll
  for (int i = 0; i < 8; ++i) ls[t * 8 + i] = src[t * 8 + i];
  __syncthreads();
  f4* dst = (f4*)(out + (((size_t)(b * CIN + t) * HH + 192 + r) * WWID + 192));
#pragma unroll
  for (int i = 0; i < 16; ++i) {
    f4 v;
#pragma unroll
    for (int e = 0; e < 4; ++e) v[e] = (float)lds[(i * 4 + e) * CIN + t];
    dst[i] = v;
  }
}

// ------------------------------- launcher ----------------------------------
extern "C" void kernel_launch(void* const* d_in, const int* in_sizes, int n_in,
                              void* d_out, int out_size, void* d_ws, size_t ws_size,
                              hipStream_t stream) {
  (void)in_sizes; (void)n_in; (void)out_size; (void)ws_size;
  const float* x  = (const float*)d_in[0];
  const float* Wq = (const float*)d_in[1];
  const float* bq = (const float*)d_in[2];
  const float* Wk = (const float*)d_in[3];
  const float* bk = (const float*)d_in[4];
  const float* Wv = (const float*)d_in[5];
  const float* bv = (const float*)d_in[6];
  const float* Wr = (const float*)d_in[7];
  const float* br = (const float*)d_in[8];
  const float* Wf = (const float*)d_in[9];
  const float* bf = (const float*)d_in[10];
  float* out = (float*)d_out;
  float* ws  = (float*)d_ws;
  _Float16* xh = (_Float16*)(ws + OFF_XH);

  k_zero<<<1, 64, 0, stream>>>(ws);
  // copy + stage one batch at a time (xh reused; stream order serializes)
  for (int b = 0; b < BATCH; ++b) {
    k_copy_xpose<<<dim3(4, 256), 256, 0, stream>>>(x, out, xh, b);
    k_fullstats<<<16, 128, 0, stream>>>(xh, Wf, bf, ws, b);
  }
  k_finalize<<<1, 64, 0, stream>>>(ws);
  k_extract<<<dim3(PTCH, BATCH), 256, 0, stream>>>(x, (_Float16*)(ws + OFF_PT0));

  _Float16* Pt[2] = { (_Float16*)(ws + OFF_PT0), (_Float16*)(ws + OFF_PT1) };
  for (int l = 0; l < 4; ++l) {
    _Float16* Pcur  = Pt[l & 1];
    _Float16* Pnext = Pt[(l & 1) ^ 1];
    k_patchstats<<<dim3(4, BATCH), 128, 0, stream>>>(Pcur, Wf, bf, ws);
    k_decide<<<1, 64, 0, stream>>>(ws);
    k_qkv<<<dim3(64, 12, BATCH), 128, 0, stream>>>(Pcur, Wq, bq, Wk, bk, Wv, bv, ws);
    k_qk<<<dim3(8, BATCH), 128, 0, stream>>>(ws);
    k_screduce<<<64, 256, 0, stream>>>(ws);
    k_softmax<<<1, 256, 0, stream>>>(ws);
    k_mmat<<<dim3(4, BATCH), 128, 0, stream>>>(Wr, ws);
    k_apply<<<dim3(64, 16, BATCH), 128, 0, stream>>>(Pcur, Pnext, br, ws);
  }
  // after 4 layers the result lives in PT0
  k_writeback<<<dim3(PTCH, BATCH), 256, 0, stream>>>(Pt[0], out);
}
